// RGCN_24215025614983
// MI455X (gfx1250) — matrix operand order repaired and checked by reading, not compiled, not run on hardware
//
#include <hip/hip_runtime.h>
#include <hip/hip_bf16.h>

typedef __attribute__((ext_vector_type(16))) __bf16 v16bf;
typedef __attribute__((ext_vector_type(8)))  float  v8f;

#define NNODES 50000
#define NREL   4
#define NEDGES 400000

// GEMM tiling: block = 256 threads = 8 wave32, C tile 128x64, K step 32
#define TM 128
#define TN 64
#define TK 32
#define LDA 40   // bf16 elems per LDS row; 80B row stride -> 8B-aligned u64 stores

union BfPair  { unsigned int u;        __bf16 h[2]; };
union BfQuad  { unsigned long long u;  __bf16 h[4]; };

// C[M,DOUT] = (A[M,256] * rs[M]) @ W[256,DOUT]  -- bf16 WMMA, f32 accumulate
template <int DOUT>
__global__ __launch_bounds__(256)
void rgcn_gemm_bf16(const float* __restrict__ A, const float* __restrict__ rs,
                    const float* __restrict__ W, float* __restrict__ C, int M)
{
    constexpr int K = 256;
    __shared__ __bf16 As[TM][LDA];   // A tile, row-major [m][k]
    __shared__ __bf16 Bs[TN][LDA];   // B tile, TRANSPOSED [n][k] so K-pairs are contiguous

    const int tid  = threadIdx.x;
    const int wave = tid >> 5;       // 0..7 -> 16-row strip
    const int lane = tid & 31;
    const int m0   = blockIdx.x * TM;
    const int n0   = blockIdx.y * TN;

    const int am = lane & 15;        // row in strip (A) / column (B, C)
    const int ah = lane >> 4;        // half selector

    // ---- k0-invariant A-tile load setup.  Rows >= M are CLAMPED to M-1
    // (branch-free: their products land in C rows >= M, which are never stored).
    int          lr[4];              // LDS row
    int          lc[4];              // LDS col (bf16 units)
    float        scA[4];             // row scale rs[row]
    const float* ap[4];              // global row pointer (+c4)
    #pragma unroll
    for (int t = 0; t < 4; ++t) {
        int f  = tid + t * 256;      // float4 index in tile (1024 total)
        int r  = f >> 3;             // row (8 float4 per 32-wide row)
        int c4 = (f & 7) << 2;
        int gr = m0 + r; if (gr > M - 1) gr = M - 1;
        lr[t]  = r;  lc[t] = c4;
        scA[t] = rs[gr];
        ap[t]  = A + (size_t)gr * K + c4;
    }

    v8f acc[4] = {};                 // 16x64 strip per wave = 4 x (16x16)

    for (int k0 = 0; k0 < K; k0 += TK) {
        // ---- A tile: 128x32 f32 -> bf16, float4 loads, u64 packed LDS stores
        #pragma unroll
        for (int t = 0; t < 4; ++t) {
            const float4 v = *(const float4*)(ap[t] + k0);
            BfQuad q;
            q.h[0] = (__bf16)(v.x * scA[t]);
            q.h[1] = (__bf16)(v.y * scA[t]);
            q.h[2] = (__bf16)(v.z * scA[t]);
            q.h[3] = (__bf16)(v.w * scA[t]);
            *(unsigned long long*)&As[lr[t]][lc[t]] = q.u;
        }
        // ---- B tile: 32x64 f32 -> bf16 transposed into [n][k]
        #pragma unroll
        for (int t = 0; t < 2; ++t) {
            int f  = tid + t * 256;          // float4 index (512 total)
            int kk = f >> 4;                 // k row (16 float4 per 64-wide row)
            int n4 = (f & 15) << 2;
            const float4 v = *(const float4*)(W + (size_t)(k0 + kk) * DOUT + n0 + n4);
            Bs[n4 + 0][kk] = (__bf16)v.x;
            Bs[n4 + 1][kk] = (__bf16)v.y;
            Bs[n4 + 2][kk] = (__bf16)v.z;
            Bs[n4 + 3][kk] = (__bf16)v.w;
        }
        __syncthreads();

        // ---- A fragment 16x32: VGPR v holds K = ((v&4)?16:0)+half*8+(v&3)*2, +1
        // (maps to 2 contiguous b128 LDS reads)
        v16bf afrag;
        const unsigned int* arow = (const unsigned int*)As[wave * 16 + am];
        #pragma unroll
        for (int vv = 0; vv < 8; ++vv) {
            int kb = ((vv & 4) << 2) + (ah << 3) + ((vv & 3) << 1);
            BfPair t; t.u = arow[kb >> 1];
            afrag[2 * vv]     = t.h[0];
            afrag[2 * vv + 1] = t.h[1];
        }

        // ---- 4 B fragments 32x16 + WMMA: VGPR v holds K = ((v&4)?16:0)+half*8+(v&3)*2, +1
        #pragma unroll
        for (int nt = 0; nt < 4; ++nt) {
            v16bf bfrag;
            const unsigned int* brow = (const unsigned int*)Bs[nt * 16 + am];
            #pragma unroll
            for (int vv = 0; vv < 8; ++vv) {
                BfPair t; t.u = brow[((vv & 4) << 1) + (ah << 2) + (vv & 3)];
                bfrag[2 * vv]     = t.h[0];
                bfrag[2 * vv + 1] = t.h[1];
            }
            acc[nt] = __builtin_amdgcn_wmma_f32_16x16x32_bf16(
                false, afrag, false, bfrag, (short)0, acc[nt], false, false);
        }
        __syncthreads();
    }

    // ---- store C: lane = N (mod 16), VGPR p -> M = p + 8*(lane>=16)
    // Compile-time DOUT => immediate-offset stores; fast path for full tiles.
    const int gm0 = m0 + wave * 16 + (ah << 3);
    float* cbase = C + (size_t)gm0 * DOUT + n0 + am;
    if (m0 + TM <= M) {
        #pragma unroll
        for (int nt = 0; nt < 4; ++nt)
            #pragma unroll
            for (int p = 0; p < 8; ++p)
                cbase[p * DOUT + nt * 16] = acc[nt][p];
    } else {
        #pragma unroll
        for (int nt = 0; nt < 4; ++nt)
            #pragma unroll
            for (int p = 0; p < 8; ++p)
                if (gm0 + p < M) cbase[p * DOUT + nt * 16] = acc[nt][p];
    }
}

// out[dst[e]] += hw[src[e]] * rinv_in[dst[e]]   (float4 rows, f32 atomics)
__global__ __launch_bounds__(256)
void rgcn_scatter(const float* __restrict__ hw, const int* __restrict__ src,
                  const int* __restrict__ dst, const float* __restrict__ rin,
                  float* __restrict__ out, int n_edges, int dout)
{
    const int tpe = dout >> 2;              // threads per edge (64 or 32)
    const int epb = 256 / tpe;              // edges per block (4 or 8)
    const int e = blockIdx.x * epb + threadIdx.x / tpe;
    if (e >= n_edges) return;
    const int c4 = (threadIdx.x % tpe) << 2;
    const int s = src[e], d = dst[e];
    const float sc = rin[d];
    const float4 v = *(const float4*)(hw + (size_t)s * dout + c4);
    float* o = out + (size_t)d * dout + c4;
    atomicAdd(o + 0, v.x * sc);
    atomicAdd(o + 1, v.y * sc);
    atomicAdd(o + 2, v.z * sc);
    atomicAdd(o + 3, v.w * sc);
}

__global__ void rgcn_zero(float* __restrict__ p, int n4)
{
    int i = blockIdx.x * blockDim.x + threadIdx.x;
    if (i < n4) ((float4*)p)[i] = make_float4(0.f, 0.f, 0.f, 0.f);
}

__global__ void rgcn_deg(const int* __restrict__ idx, float* __restrict__ deg, int n)
{
    int i = blockIdx.x * blockDim.x + threadIdx.x;
    if (i < n) atomicAdd(deg + idx[i], 1.0f);
}

__global__ void rgcn_rsqrt(float* __restrict__ p, int n)
{
    int i = blockIdx.x * blockDim.x + threadIdx.x;
    if (i < n) p[i] = rsqrtf(fmaxf(p[i], 1.0f));
}

// o = (relu?) (acc + sum_r b[r])   b is [4, dout]; float4 over total elements
__global__ void rgcn_bias_act(const float* __restrict__ acc, const float* __restrict__ b,
                              float* __restrict__ o, int total4, int dout, int relu)
{
    int i = blockIdx.x * blockDim.x + threadIdx.x;
    if (i >= total4) return;
    int c = (i % (dout >> 2)) << 2;
    float4 v = ((const float4*)acc)[i];
    #pragma unroll
    for (int r = 0; r < NREL; ++r) {
        const float4 bb = *(const float4*)(b + r * dout + c);
        v.x += bb.x; v.y += bb.y; v.z += bb.z; v.w += bb.w;
    }
    if (relu) {
        v.x = fmaxf(v.x, 0.f); v.y = fmaxf(v.y, 0.f);
        v.z = fmaxf(v.z, 0.f); v.w = fmaxf(v.w, 0.f);
    }
    ((float4*)o)[i] = v;
}

extern "C" void kernel_launch(void* const* d_in, const int* in_sizes, int n_in,
                              void* d_out, int out_size, void* d_ws, size_t ws_size,
                              hipStream_t stream)
{
    const float* x = (const float*)d_in[0];
    const float* Wl[4] = { (const float*)d_in[1], (const float*)d_in[3],
                           (const float*)d_in[5], (const float*)d_in[7] };
    const float* bl[4] = { (const float*)d_in[2], (const float*)d_in[4],
                           (const float*)d_in[6], (const float*)d_in[8] };
    const int* src = (const int*)d_in[9];
    const int* dst = (const int*)d_in[10];

    float* ws       = (float*)d_ws;
    float* rinv_out = ws;                                   // [4][N]
    float* rinv_in  = rinv_out + NREL * NNODES;             // [4][N]
    float* HW       = rinv_in  + NREL * NNODES;             // [N][256]
    float* OUT      = HW  + (size_t)NNODES * 256;           // [N][256]
    float* H        = OUT + (size_t)NNODES * 256;           // [N][256]

    // ---- degrees (once; identical for every layer), then rsqrt(clip(deg,1))
    {
        int n = 2 * NREL * NNODES;
        rgcn_zero<<<(n / 4 + 255) / 256, 256, 0, stream>>>(rinv_out, n / 4);
        for (int r = 0; r < NREL; ++r) {
            rgcn_deg<<<(NEDGES + 255) / 256, 256, 0, stream>>>(
                src + (size_t)r * NEDGES, rinv_out + r * NNODES, NEDGES);
            rgcn_deg<<<(NEDGES + 255) / 256, 256, 0, stream>>>(
                dst + (size_t)r * NEDGES, rinv_in + r * NNODES, NEDGES);
        }
        rgcn_rsqrt<<<(n + 255) / 256, 256, 0, stream>>>(rinv_out, n);
    }

    // ---- 4 layers
    const float* hin = x;
    for (int l = 0; l < 4; ++l) {
        const int dout  = (l == 3) ? 128 : 256;
        const int din   = 256;
        const int total = NNODES * dout;

        rgcn_zero<<<(total / 4 + 255) / 256, 256, 0, stream>>>(OUT, total / 4);

        for (int r = 0; r < NREL; ++r) {
            dim3 g((NNODES + TM - 1) / TM, dout / TN);
            if (dout == 256)
                rgcn_gemm_bf16<256><<<g, 256, 0, stream>>>(
                    hin, rinv_out + r * NNODES,
                    Wl[l] + (size_t)r * din * dout, HW, NNODES);
            else
                rgcn_gemm_bf16<128><<<g, 256, 0, stream>>>(
                    hin, rinv_out + r * NNODES,
                    Wl[l] + (size_t)r * din * dout, HW, NNODES);

            const int epb = 256 / (dout >> 2);
            rgcn_scatter<<<(NEDGES + epb - 1) / epb, 256, 0, stream>>>(
                HW, src + (size_t)r * NEDGES, dst + (size_t)r * NEDGES,
                rinv_in + r * NNODES, OUT, NEDGES, dout);
        }

        float* hnext = (l == 3) ? (float*)d_out : H;
        rgcn_bias_act<<<(total / 4 + 255) / 256, 256, 0, stream>>>(
            OUT, bl[l], hnext, total / 4, dout, (l < 3) ? 1 : 0);
        hin = hnext;
    }
}
